// SoftMaxAttentionAggregation_25202868093636
// MI455X (gfx1250) — matrix operand order repaired and checked by reading, not compiled, not run on hardware
//
#include <hip/hip_runtime.h>

#define DIM 128
#define NNODES 1000000
#define NGRAPHS 4096
#define NTILES (NNODES / 16)   // 62500 exact
#define WSTR 136               // padded LDS row stride in 16-bit elems (272B = 17*16B)

typedef __attribute__((ext_vector_type(16))) __bf16          v16bf;
typedef __attribute__((ext_vector_type(16))) unsigned short  v16s;
typedef __attribute__((ext_vector_type(8)))  unsigned short  v8s;
typedef __attribute__((ext_vector_type(8)))  float           v8f;

// ---- helpers -------------------------------------------------------------

// native f32 -> bf16 convert (lets clang emit v_cvt_pk_bf16_f32 for pairs)
__device__ __forceinline__ unsigned short f2bf(float f) {
  return __builtin_bit_cast(unsigned short, (__bf16)f);
}
__device__ __forceinline__ unsigned pack2bf(float lo, float hi) {
  return (unsigned)f2bf(lo) | ((unsigned)f2bf(hi) << 16);
}

// order-preserving float<->uint encoding for atomicMax on floats of any sign
__device__ __forceinline__ unsigned encf(float f) {
  unsigned u = __float_as_uint(f);
  return (u & 0x80000000u) ? ~u : (u | 0x80000000u);
}
__device__ __forceinline__ float decf(unsigned u) {
  u = (u & 0x80000000u) ? (u & 0x7fffffffu) : ~u;
  return __uint_as_float(u);
}

// build a 16-elem bf16 fragment from two 16B LDS chunks (2x ds_load_b128)
__device__ __forceinline__ v16bf frag2(const unsigned short* p0, const unsigned short* p1) {
  v8s a = *(const v8s*)p0;
  v8s b = *(const v8s*)p1;
  v16s t;
#pragma unroll
  for (int i = 0; i < 8; ++i) { t[i] = a[i]; t[i + 8] = b[i]; }
  return __builtin_bit_cast(v16bf, t);
}

// stage a 128x128 f32 weight matrix as bf16, TRANSPOSED: WT[n*WSTR + k] = W[k][n]
__device__ __forceinline__ void stage_wt(const float* __restrict__ W, unsigned short* WT, int tid) {
  for (int i = tid; i < DIM * DIM; i += 256) {
    int k = i >> 7, n = i & 127;          // coalesced global read along n
    WT[n * WSTR + k] = f2bf(W[i]);
  }
}

// stage one 16x128 f32 x-tile as bf16, row-major with padded stride (wave-local)
__device__ __forceinline__ void stage_x(const float* __restrict__ x, unsigned short* XS,
                                        int lane, long tb) {
#pragma unroll 4
  for (int r = 0; r < 16; ++r) {
    float4 v = ((const float4*)(x + (size_t)(tb + r) * DIM))[lane];
    uint2 p;
    p.x = pack2bf(v.x, v.y);
    p.y = pack2bf(v.z, v.w);
    *(uint2*)(XS + r * WSTR + lane * 4) = p;
  }
}

// prefetch next tile's 8KB of x into cache (global_prefetch_b8; no counter cost)
__device__ __forceinline__ void prefetch_x(const float* __restrict__ x, int lane, long tb_next) {
  const float* p = x + (size_t)tb_next * DIM + (size_t)lane * 64;  // 256B per lane
  __builtin_prefetch(p, 0, 3);
  __builtin_prefetch(p + 32, 0, 3);                                // +128B
}

// 16x128 (A, LDS) @ 128x128 (B=WT transposed, LDS) -> 16x128 f32 acc via WMMA
// A layout per ISA: lanes 0-15 row M=lane, K octets {kb..kb+7},{kb+16..kb+23};
//                   lanes 16-31 same rows, K {kb+8..},{kb+24..}
// B layout: lanes 0-15 col N=lane, K octets {kb..kb+7},{kb+16..kb+23}; lanes 16-31 K {kb+8..},{kb+24..}
__device__ __forceinline__ void gemm_tile(const unsigned short* AS, const unsigned short* WT,
                                          int lane, v8f acc[8]) {
  const int half = lane >> 4, lr = lane & 15;
#pragma unroll
  for (int k = 0; k < 4; ++k) {
    const int kb = k * 32;
    const unsigned short* ap = AS + lr * WSTR + kb + half * 8;
    v16bf A = frag2(ap, ap + 16);
#pragma unroll
    for (int n = 0; n < 8; ++n) {
      const unsigned short* bp = WT + (n * 16 + lr) * WSTR + kb + half * 8;
      v16bf B = frag2(bp, bp + 16);
      acc[n] = __builtin_amdgcn_wmma_f32_16x16x32_bf16(
          false, A, false, B, (short)0, acc[n], false, false);
    }
  }
}

// ---- kernels -------------------------------------------------------------

__global__ void __launch_bounds__(256)
init_kernel(float* __restrict__ out, unsigned* __restrict__ smax, float* __restrict__ ssum) {
  int t = blockIdx.x * 256 + threadIdx.x;
  if (t < NGRAPHS * DIM) out[t] = 0.f;
  if (t < NGRAPHS) { smax[t] = 0x007fffffu /* enc(-inf) */; ssum[t] = 0.f; }
}

__global__ void __launch_bounds__(256)
gate_kernel(const float* __restrict__ x, const int* __restrict__ idx,
            const float* __restrict__ Wg1, const float* __restrict__ bg1,
            const float* __restrict__ Wg2, const float* __restrict__ bg2,
            float* __restrict__ g, unsigned* __restrict__ smax) {
  __shared__ unsigned short WT[DIM * WSTR];
  __shared__ unsigned short XS[8][16 * WSTR];
  __shared__ float BG1[DIM], WG2v[DIM];

  const int tid = threadIdx.x, lane = tid & 31, w = tid >> 5;
  const int half = lane >> 4, lr = lane & 15;
  const int stride = gridDim.x * 8;

  stage_wt(Wg1, WT, tid);
  if (tid < DIM) { BG1[tid] = bg1[tid]; WG2v[tid] = Wg2[tid]; }
  __syncthreads();
  const float bg2v = bg2[0];

  for (int t = blockIdx.x * 8 + w; t < NTILES; t += stride) {
    const long tb = (long)t * 16;
    if (t + stride < NTILES) prefetch_x(x, lane, (long)(t + stride) * 16);
    stage_x(x, XS[w], lane, tb);

    v8f acc[8] = {};
    gemm_tile(XS[w], WT, lane, acc);

    float part[8];
#pragma unroll
    for (int r = 0; r < 8; ++r) part[r] = 0.f;
#pragma unroll
    for (int n = 0; n < 8; ++n) {
      const int col = n * 16 + lr;
      const float b = BG1[col], w2 = WG2v[col];
#pragma unroll
      for (int r = 0; r < 8; ++r) {
        float v = acc[n][r] + b;
        float ys = v / (1.f + __expf(-v));   // SiLU
        part[r] += ys * w2;
      }
    }
    // reduce across the 16 lanes of each half (columns) -> per-row gate value
#pragma unroll
    for (int m = 8; m >= 1; m >>= 1) {
#pragma unroll
      for (int r = 0; r < 8; ++r) part[r] += __shfl_xor(part[r], m, 32);
    }
    if (lr == 0) {
      const long mb = tb + half * 8;
#pragma unroll
      for (int r = 0; r < 8; ++r) {
        float gv = part[r] + bg2v;
        g[mb + r] = gv;
        atomicMax(&smax[idx[mb + r]], encf(gv));
      }
    }
  }
}

__global__ void __launch_bounds__(256)
norm_kernel(float* __restrict__ g, const int* __restrict__ idx,
            const unsigned* __restrict__ smax, float* __restrict__ ssum) {
  int t = blockIdx.x * 256 + threadIdx.x;
  if (t < NNODES) {
    int seg = idx[t];
    float e = __expf(g[t] - decf(smax[seg]));
    g[t] = e;                       // overwrite gate with exp-shifted value
    atomicAdd(&ssum[seg], e);
  }
}

__global__ void __launch_bounds__(256)
node_kernel(const float* __restrict__ x, const int* __restrict__ idx,
            const float* __restrict__ Wn1, const float* __restrict__ bn1,
            const float* __restrict__ Wn2, const float* __restrict__ bn2,
            const float* __restrict__ e, const float* __restrict__ ssum,
            float* __restrict__ out) {
  __shared__ unsigned short WT1[DIM * WSTR];
  __shared__ unsigned short WT2[DIM * WSTR];
  __shared__ unsigned short XS[8][16 * WSTR];   // x tile, then reused for y tile
  __shared__ float BN1[DIM], BN2[DIM];

  const int tid = threadIdx.x, lane = tid & 31, w = tid >> 5;
  const int half = lane >> 4, lr = lane & 15;
  const int stride = gridDim.x * 8;

  stage_wt(Wn1, WT1, tid);
  stage_wt(Wn2, WT2, tid);
  if (tid < DIM) { BN1[tid] = bn1[tid]; BN2[tid] = bn2[tid]; }
  __syncthreads();

  for (int t = blockIdx.x * 8 + w; t < NTILES; t += stride) {
    const long tb = (long)t * 16;
    if (t + stride < NTILES) prefetch_x(x, lane, (long)(t + stride) * 16);
    stage_x(x, XS[w], lane, tb);

    v8f acc[8] = {};
    gemm_tile(XS[w], WT1, lane, acc);

    // SiLU + write y (bf16) back over this wave's LDS region (C-layout -> row-major)
#pragma unroll
    for (int n = 0; n < 8; ++n) {
      const int col = n * 16 + lr;
      const float b = BN1[col];
#pragma unroll
      for (int r = 0; r < 8; ++r) {
        float v = acc[n][r] + b;
        float ys = v / (1.f + __expf(-v));
        XS[w][(r + half * 8) * WSTR + col] = f2bf(ys);
      }
    }

    v8f acc2[8] = {};
    gemm_tile(XS[w], WT2, lane, acc2);   // per-wave DS ops are in-order: stores precede loads

    // attention coefficient for the 16 rows of this tile
    const long node = tb + lr;
    const float ev = e[node];
    const int seg = idx[node];
    const float cl = ev / ssum[seg];

    float cr[8]; int sr[8];
#pragma unroll
    for (int r = 0; r < 8; ++r) {
      cr[r] = __shfl(cl, half * 8 + r, 32);
      sr[r] = __shfl(seg, half * 8 + r, 32);
    }

    // run-compacted segment-sum into out (index is sorted)
#pragma unroll
    for (int n = 0; n < 8; ++n) {
      const int col = n * 16 + lr;
      const float b2 = BN2[col];
      float run = 0.f;
      int prev = sr[0];
#pragma unroll
      for (int r = 0; r < 8; ++r) {
        float v = cr[r] * (acc2[n][r] + b2);
        if (r && sr[r] != prev) {
          atomicAdd(&out[(size_t)prev * DIM + col], run);
          run = 0.f;
          prev = sr[r];
        }
        run += v;
      }
      atomicAdd(&out[(size_t)prev * DIM + col], run);
    }
  }
}

// ---- launch --------------------------------------------------------------

extern "C" void kernel_launch(void* const* d_in, const int* in_sizes, int n_in,
                              void* d_out, int out_size, void* d_ws, size_t ws_size,
                              hipStream_t stream) {
  const float* x   = (const float*)d_in[0];
  const int*   idx = (const int*)d_in[1];
  // d_in[2] = dim_size scalar (compile-time NGRAPHS)
  const float* Wg1 = (const float*)d_in[3];
  const float* bg1 = (const float*)d_in[4];
  const float* Wg2 = (const float*)d_in[5];
  const float* bg2 = (const float*)d_in[6];
  const float* Wn1 = (const float*)d_in[7];
  const float* bn1 = (const float*)d_in[8];
  const float* Wn2 = (const float*)d_in[9];
  const float* bn2 = (const float*)d_in[10];
  float* out = (float*)d_out;

  char* ws = (char*)d_ws;
  float*    g    = (float*)ws;                                            // 1M floats (gate -> e)
  unsigned* smax = (unsigned*)(ws + (size_t)NNODES * sizeof(float));      // 4096 u32
  float*    ssum = (float*)(ws + (size_t)NNODES * sizeof(float)
                               + (size_t)NGRAPHS * sizeof(unsigned));     // 4096 f32

  init_kernel<<<(NGRAPHS * DIM + 255) / 256, 256, 0, stream>>>(out, smax, ssum);
  gate_kernel<<<2048, 256, 0, stream>>>(x, idx, Wg1, bg1, Wg2, bg2, g, smax);
  norm_kernel<<<(NNODES + 255) / 256, 256, 0, stream>>>(g, idx, smax, ssum);
  node_kernel<<<2048, 256, 0, stream>>>(x, idx, Wn1, bn1, Wn2, bn2, g, ssum, out);
}
